// Tree_81844896793183
// MI455X (gfx1250) — compile-verified, matrix operand order checked by reading, not run on hardware
//
#include <hip/hip_runtime.h>
#include <hip/hip_bf16.h>

// ---------------------------------------------------------------------------
// Hierarchical octree builder for MI455X (gfx1250, wave32, WGP).
// Segment aggregation runs on the matrix pipe: one-hot f16 A-matrix times
// hi/lo-split f16 feature B-matrix, f32 accumulate via v_wmma_f32_16x16x32_f16.
// LDS staging is kept in WMMA *fragment order* so fragment assembly is pure
// ds_load_b128 traffic.
// ---------------------------------------------------------------------------

typedef __attribute__((ext_vector_type(16))) _Float16 v16h;
typedef __attribute__((ext_vector_type(8)))  float    v8f;

#define TPB 256

// ----------------------------- utility kernels -----------------------------

__global__ void k_zero_i32(int* __restrict__ p, long n) {
  long i = (long)blockIdx.x * TPB + threadIdx.x;
  long stride = (long)gridDim.x * TPB;
  for (; i < n; i += stride) p[i] = 0;
}

__global__ void k_zero_f32(float* __restrict__ p, long n) {
  long i = (long)blockIdx.x * TPB + threadIdx.x;
  long stride = (long)gridDim.x * TPB;
  for (; i < n; i += stride) p[i] = 0.0f;
}

__global__ void k_seti(int* p, int v) {
  if (threadIdx.x == 0 && blockIdx.x == 0) *p = v;
}

__global__ void k_bump(int* c, const int* nPtr, int mult) {
  if (threadIdx.x == 0 && blockIdx.x == 0) *c += mult * (*nPtr);
}

// ----------------------------- bounds & binning ----------------------------

__global__ void k_init_bounds(int* pi, float* pf) {
  int t = threadIdx.x;
  if (blockIdx.x == 0 && t < 3) {
    pi[t]     = 0x7F800000;  // +inf bits (positions are non-negative)
    pi[3 + t] = 0;           // 0.0f bits
    pf[6 + t] = 0.0f;        // sum
  }
}

__global__ void k_bounds(const float* __restrict__ nodes, int N, int* pi, float* pf) {
  __shared__ float smn[3][TPB], smx[3][TPB], ssm[3][TPB];
  int i = blockIdx.x * TPB + threadIdx.x;
  float mn[3] = {1e30f, 1e30f, 1e30f};
  float mx[3] = {-1e30f, -1e30f, -1e30f};
  float sm[3] = {0.f, 0.f, 0.f};
  if (i < N) {
    #pragma unroll
    for (int d = 0; d < 3; d++) {
      float v = nodes[(long)i * 10 + 4 + d];   // pos = nd[:, -6:-3]
      mn[d] = v; mx[d] = v; sm[d] = v;
    }
  }
  #pragma unroll
  for (int d = 0; d < 3; d++) { smn[d][threadIdx.x] = mn[d]; smx[d][threadIdx.x] = mx[d]; ssm[d][threadIdx.x] = sm[d]; }
  __syncthreads();
  for (int off = TPB / 2; off > 0; off >>= 1) {
    if (threadIdx.x < off) {
      #pragma unroll
      for (int d = 0; d < 3; d++) {
        smn[d][threadIdx.x] = fminf(smn[d][threadIdx.x], smn[d][threadIdx.x + off]);
        smx[d][threadIdx.x] = fmaxf(smx[d][threadIdx.x], smx[d][threadIdx.x + off]);
        ssm[d][threadIdx.x] += ssm[d][threadIdx.x + off];
      }
    }
    __syncthreads();
  }
  if (threadIdx.x == 0) {
    #pragma unroll
    for (int d = 0; d < 3; d++) {
      atomicMin(&pi[d], __float_as_int(smn[d][0]));      // valid for non-neg floats
      atomicMax(&pi[3 + d], __float_as_int(smx[d][0]));
      atomicAdd(&pf[6 + d], ssm[d][0]);
    }
  }
}

__global__ void k_params(const int* pi, float* pf, int N, int rl) {
  if (threadIdx.x == 0 && blockIdx.x == 0) {
    float box = 0.f;
    for (int d = 0; d < 3; d++) {
      float mn = __int_as_float(pi[d]), mx = __int_as_float(pi[3 + d]);
      box = fmaxf(box, mx - mn);
    }
    float invN = 1.0f / (float)N;
    for (int d = 0; d < 3; d++) {
      float ref = pf[6 + d] * invN;
      pf[9 + d] = ref - box * 0.5f;                      // lo_d
    }
    pf[12] = (box > 0.f) ? ((float)rl / box) : 0.f;      // 1/step
  }
}

__global__ void k_assign(const float* __restrict__ nodes, const float* __restrict__ pf,
                         int* __restrict__ cellid, int N, int rl) {
  int i = blockIdx.x * TPB + threadIdx.x;
  if (i >= N) return;
  float inv = pf[12];
  int g[3];
  #pragma unroll
  for (int d = 0; d < 3; d++) {
    float v = nodes[(long)i * 10 + 4 + d];
    int t = (int)floorf((v - pf[9 + d]) * inv);
    g[d] = min(max(t, 0), rl - 1);
  }
  int xg = g[0];
  int yg = rl - 1 - g[1];
  int zg = rl - 1 - g[2];
  cellid[i] = xg + yg * rl + zg * rl * rl;
}

// -------------------------- histogram / scan / compaction ------------------

__global__ void k_hist(const int* __restrict__ ids, const int* nPtr, int* __restrict__ counts) {
  int n = *nPtr;
  int i = blockIdx.x * TPB + threadIdx.x;
  if (i < n) atomicAdd(&counts[ids[i]], 1);
}

__global__ void k_scan_block(const int* __restrict__ in, int mode,
                             int* __restrict__ out, int* __restrict__ sums, int M) {
  __shared__ int tmp[TPB];
  int i = blockIdx.x * TPB + threadIdx.x;
  int v = 0;
  if (i < M) { int x = in[i]; v = mode ? (x > 0 ? 1 : 0) : x; }
  tmp[threadIdx.x] = v;
  __syncthreads();
  for (int off = 1; off < TPB; off <<= 1) {
    int t = (threadIdx.x >= off) ? tmp[threadIdx.x - off] : 0;
    __syncthreads();
    tmp[threadIdx.x] += t;
    __syncthreads();
  }
  if (i < M) out[i] = tmp[threadIdx.x] - v;  // exclusive
  if (threadIdx.x == TPB - 1) sums[blockIdx.x] = tmp[TPB - 1];
}

__global__ void k_scan_sums(int* sums, int nb, int* total) {
  if (threadIdx.x == 0 && blockIdx.x == 0) {
    int acc = 0;
    for (int b = 0; b < nb; b++) { int t = sums[b]; sums[b] = acc; acc += t; }
    if (total) *total = acc;
  }
}

__global__ void k_scan_add(int* __restrict__ out, const int* __restrict__ sums, int M) {
  int i = blockIdx.x * TPB + threadIdx.x;
  if (i < M) out[i] += sums[blockIdx.x];
}

__global__ void k_compact(const int* __restrict__ counts, const int* __restrict__ flagscan,
                          int M, int* __restrict__ ne, int* __restrict__ d2c) {
  int i = blockIdx.x * TPB + threadIdx.x;
  if (i >= M) return;
  if (counts[i] > 0) { ne[flagscan[i]] = i; d2c[i] = flagscan[i]; }
  else d2c[i] = -1;
}

__global__ void k_scatter(const int* __restrict__ ids, const int* nPtr,
                          const int* __restrict__ starts, int* __restrict__ cursor,
                          const int* __restrict__ d2c, int* __restrict__ order,
                          int* __restrict__ segc) {
  int n = *nPtr;
  int i = blockIdx.x * TPB + threadIdx.x;
  if (i >= n) return;
  int d = ids[i];
  int pos = starts[d] + atomicAdd(&cursor[d], 1);
  order[pos] = i;
  segc[i] = d2c[d];
}

// --------------------- WMMA segment aggregation (the math) -----------------
//
// Rows are pre-sorted by segment (gathered through `order`), so 32 consecutive
// rows span <= 32 contiguous compact segments.  Per 32-row chunk and per wave:
//   A1/A2 = 16x32 one-hot (exact in f16):  A[r][k] = (seg_k - seg_base == r)
//   B     = 32x16 features (col0 = mass, col c = feat_c * mass, cols>=10 = 0),
//           split x = hi + lo for f32-exact accumulation.
//   D += A1*Bhi + A1*Blo ; D2 += A2*Bhi + A2*Blo     (v_wmma_f32_16x16x32_f16)
// LDS staging uses WMMA fragment order: the per-lane fragment element sequence
// (ISA 7.12.2, 16-bit A/B layout) visits source lanes k in order
// {0..7,16..23 | 8..15,24..31}, i.e. lane index with bits 3<->4 swapped.
// Writing at pos = swap34(lane) makes each consumer's 16 f16 elements a single
// contiguous 32B run -> two ds_load_b128 per fragment, and the one-hot rows
// come from one ds_load_b128 of packed rloc bytes.

__global__ __launch_bounds__(TPB) void k_agg_wmma(
    const float* __restrict__ src, int srcStride,
    const int* __restrict__ order, const int* __restrict__ seg,
    const int* nInPtr, float* __restrict__ svOut, const int* nOutPtr) {
  __shared__ alignas(64) _Float16      sHiT[8][10][32];  // [wave][col][fragment pos]
  __shared__ alignas(64) _Float16      sLoT[8][10][32];
  __shared__ alignas(16) unsigned char sRb[8][32];       // rloc bytes, fragment order

  const int nIn  = *nInPtr;
  const int nOut = *nOutPtr;
  const int wave = threadIdx.x >> 5;
  const int lane = threadIdx.x & 31;
  const int mRow = lane & 15;
  const int half = lane >> 4;
  const int pos  = (lane & ~24) | ((lane & 8) << 1) | ((lane & 16) >> 1);  // swap bits 3<->4
  const int totalChunks = (nIn + 31) / 32;

  for (int chunk = blockIdx.x * 8 + wave; chunk < totalChunks; chunk += gridDim.x * 8) {
    int j = chunk * 32 + lane;
    int r = 0x7fffffff;
    float vals[10];
    #pragma unroll
    for (int c = 0; c < 10; c++) vals[c] = 0.f;
    if (j < nIn) {
      int row = order[j];
      r = seg[row];
      const float* p = src + (long)row * srcStride;
      float mass = p[0];
      vals[0] = mass;
      #pragma unroll
      for (int c = 1; c < 10; c++) vals[c] = p[c] * mass;   // mass-weighted
    }
    int sBase = __builtin_amdgcn_readfirstlane(r);          // lane0 is always valid
    sRb[wave][pos] = (unsigned char)((r == 0x7fffffff) ? 255 : (r - sBase));
    #pragma unroll
    for (int c = 0; c < 10; c++) {
      _Float16 h = (_Float16)vals[c];
      sHiT[wave][c][pos] = h;
      sLoT[wave][c][pos] = (_Float16)(vals[c] - (float)h);
    }
    // wave-private LDS region: same-wave DS ordering + compiler dscnt waits
    // make the data visible; no workgroup barrier needed.

    // A fragments: one b128 load of 16 packed rloc bytes, byte-extract compares.
    int4 rb = *(const int4*)&sRb[wave][half * 16];
    int rw[4] = {rb.x, rb.y, rb.z, rb.w};
    v16h a1, a2;
    #pragma unroll
    for (int i = 0; i < 16; i++) {
      int rk = (rw[i >> 2] >> ((i & 3) * 8)) & 255;
      a1[i] = (rk == mRow)      ? (_Float16)1.0f : (_Float16)0.0f;
      a2[i] = (rk == mRow + 16) ? (_Float16)1.0f : (_Float16)0.0f;
    }
    // B fragments: 32 contiguous bytes -> two ds_load_b128 each.
    // Columns 10..15 are structurally zero and never touch LDS.
    v16h bh = {}, bl = {};
    if (mRow < 10) {
      bh = *(const v16h*)&sHiT[wave][mRow][half * 16];
      bl = *(const v16h*)&sLoT[wave][mRow][half * 16];
    }

    v8f c1 = {}, c2 = {};
    c1 = __builtin_amdgcn_wmma_f32_16x16x32_f16(false, a1, false, bh, (short)0, c1, false, false);
    c1 = __builtin_amdgcn_wmma_f32_16x16x32_f16(false, a1, false, bl, (short)0, c1, false, false);
    c2 = __builtin_amdgcn_wmma_f32_16x16x32_f16(false, a2, false, bh, (short)0, c2, false, false);
    c2 = __builtin_amdgcn_wmma_f32_16x16x32_f16(false, a2, false, bl, (short)0, c2, false, false);

    int col = lane & 15;
    #pragma unroll
    for (int vv = 0; vv < 8; vv++) {
      int row1 = sBase + vv + 8 * half;          // C/D layout: M = vgpr + 8*(lane>=16)
      float x = c1[vv];
      if (x != 0.f && row1 < nOut) atomicAdd(&svOut[(long)row1 * 16 + col], x);
      int row2 = sBase + 16 + vv + 8 * half;
      x = c2[vv];
      if (x != 0.f && row2 < nOut) atomicAdd(&svOut[(long)row2 * 16 + col], x);
    }
  }
}

__global__ void k_finalize_sv(float* __restrict__ sv, const int* nPtr) {
  int n = *nPtr;
  int i = blockIdx.x * TPB + threadIdx.x;
  if (i >= n) return;
  float m = sv[(long)i * 16];
  float inv = (m != 0.f) ? 1.0f / m : 0.f;
  #pragma unroll
  for (int c = 1; c < 10; c++) sv[(long)i * 16 + c] *= inv;   // weighted mean
  #pragma unroll
  for (int c = 10; c < 16; c++) sv[(long)i * 16 + c] = 0.f;
}

// ------------------------------ output emitters ----------------------------

__global__ void k_assigns_out(const int* __restrict__ order, const int* __restrict__ segc,
                              const int* nPtr, int* __restrict__ outp,
                              const int* offPtr, long base, long capEnd) {
  int n = *nPtr;
  int j = blockIdx.x * TPB + threadIdx.x;
  if (j >= n) return;
  long off = base + *offPtr;
  if (off + n + j < capEnd) {
    int oj = order[j];
    outp[off + j]     = segc[oj];   // seg0[order]
    outp[off + n + j] = oj;         // order
  }
}

__global__ void k_sv_out(const float* __restrict__ sv, const int* nPtr,
                         float* __restrict__ outf, const int* offPtr, long base, long capEnd) {
  int n = *nPtr;
  int i = blockIdx.x * TPB + threadIdx.x;
  if (i >= n * 10) return;
  int row = i / 10, c = i % 10;
  long off = base + *offPtr + i;
  if (off < capEnd) outf[off] = sv[(long)row * 16 + c];
}

__global__ void k_ids_out(const int* __restrict__ ne, const int* nPtr,
                          int* __restrict__ outp, const int* offPtr, long base, long capEnd) {
  int n = *nPtr;
  int i = blockIdx.x * TPB + threadIdx.x;
  if (i >= n) return;
  long off = base + *offPtr + i;
  if (off < capEnd) outp[off] = ne[i];
}

__global__ void k_parent(const int* __restrict__ ne, const int* nPtr, int rl,
                         int* __restrict__ childIds) {
  int n = *nPtr;
  int i = blockIdx.x * TPB + threadIdx.x;
  if (i >= n) return;
  int id = ne[i];
  int x = id % rl, y = (id / rl) % rl, z = id / (rl * rl);
  int crl = rl >> 1;
  childIds[i] = (x >> 1) + (y >> 1) * crl + (z >> 1) * crl * crl;  // _build_clusters parent
}

__global__ void k_graph(const int* __restrict__ cellid, const int* __restrict__ counts,
                        const int* __restrict__ starts, const int* __restrict__ order,
                        int N, int rl, int* __restrict__ outp, long capEnd,
                        unsigned int* cursor) {
  int i = blockIdx.x * TPB + threadIdx.x;
  if (i >= N) return;
  int id = cellid[i];
  int x = id % rl, y = (id / rl) % rl, z = id / (rl * rl);
  for (int t = 0; t < 27; t++) {
    int nx = x + t % 3 - 1, ny = y + (t / 3) % 3 - 1, nz = z + t / 9 - 1;
    if ((unsigned)nx >= (unsigned)rl || (unsigned)ny >= (unsigned)rl || (unsigned)nz >= (unsigned)rl) continue;
    int nd = nx + ny * rl + nz * rl * rl;
    int c = counts[nd];
    if (!c) continue;
    int s = starts[nd];
    __builtin_prefetch(&order[s], 0, 1);                     // global_prefetch_b8
    unsigned base = atomicAdd(cursor, (unsigned)c);
    for (int k = 0; k < c; k++) {
      long p = (long)(base + k) * 2;
      if (p + 1 < capEnd) { outp[p] = order[s + k]; outp[p + 1] = i; }  // (dst, src)
    }
  }
}

__global__ void k_edges(const int* __restrict__ ne, const int* nPtr,
                        const int* __restrict__ counts, const int* __restrict__ d2c,
                        int rl, int* __restrict__ outp, long base, long capEnd,
                        unsigned int* cursor) {
  int n = *nPtr;
  int i = blockIdx.x * TPB + threadIdx.x;
  if (i >= n) return;
  int id = ne[i];
  int x = id % rl, y = (id / rl) % rl, z = id / (rl * rl);
  for (int t = 0; t < 27; t++) {
    int nx = x + t % 3 - 1, ny = y + (t / 3) % 3 - 1, nz = z + t / 9 - 1;
    if ((unsigned)nx >= (unsigned)rl || (unsigned)ny >= (unsigned)rl || (unsigned)nz >= (unsigned)rl) continue;
    int nd = nx + ny * rl + nz * rl * rl;
    if (counts[nd] <= 0) continue;
    unsigned p = atomicAdd(cursor, 1u);
    long off = base + (long)p * 2;
    if (off + 1 < capEnd) { outp[off] = d2c[nd]; outp[off + 1] = i; }
  }
}

// ------------------------------- host driver -------------------------------

extern "C" void kernel_launch(void* const* d_in, const int* in_sizes, int n_in,
                              void* d_out, int out_size, void* d_ws, size_t ws_size,
                              hipStream_t stream) {
  const float* nodes = (const float*)d_in[0];
  const int N = in_sizes[0] / 10;
  const int GRID0 = 262144;                  // 64^3 dense cells at level 0

  char* w = (char*)d_ws;
  size_t o = 0;
  auto A = [&](size_t b) { size_t r = o; o = (o + b + 255) & ~(size_t)255; return r; };
  size_t oParams = A(64 * 4),  oCnt  = A(64 * 4);
  size_t oCell  = A((size_t)N * 4), oChild = A((size_t)N * 4);
  size_t oOrder = A((size_t)N * 4), oSeg   = A((size_t)N * 4);
  size_t oCounts = A((size_t)GRID0 * 4), oStarts = A((size_t)GRID0 * 4);
  size_t oFlag   = A((size_t)GRID0 * 4), oCur    = A((size_t)GRID0 * 4);
  size_t oSums   = A(2048 * 4);
  size_t oNe     = A((size_t)GRID0 * 4), oD2C = A((size_t)GRID0 * 4);
  size_t oSvA    = A((size_t)GRID0 * 16 * 4);      // 16 MB
  size_t oSvB    = A((size_t)32768 * 16 * 4);      //  2 MB
  if (o > ws_size) return;                          // deterministic no-op if ws too small

  int*   pi    = (int*)(w + oParams);
  float* pf    = (float*)(w + oParams);
  int*   cnt   = (int*)(w + oCnt);       // [0..4]=n_level, 5=graph, 6=assignOff, 7=svOff, 8=idsOff, 9=edgeOff, 10=N
  int*   cell  = (int*)(w + oCell);
  int*   child = (int*)(w + oChild);
  int*   order = (int*)(w + oOrder);
  int*   segc  = (int*)(w + oSeg);
  int*   counts = (int*)(w + oCounts);
  int*   starts = (int*)(w + oStarts);
  int*   flagsc = (int*)(w + oFlag);
  int*   cursor = (int*)(w + oCur);
  int*   sums   = (int*)(w + oSums);
  int*   ne     = (int*)(w + oNe);
  int*   d2c    = (int*)(w + oD2C);
  float* svA    = (float*)(w + oSvA);
  float* svB    = (float*)(w + oSvB);

  int*   outI = (int*)d_out;
  float* outF = (float*)d_out;
  long OS = out_size;
  long gGraphEnd = OS / 2;                                  // graph pairs at [0, OS/2)
  long gABase = OS / 2,       gAEnd = (OS * 5) / 8;         // assigns
  long gSBase = (OS * 5) / 8, gSEnd = (OS * 7) / 8;         // svs (floats)
  long gEBase = (OS * 7) / 8, gEEnd = (OS * 15) / 16;       // edges
  long gIBase = (OS * 15) / 16, gIEnd = OS;                 // ids

  auto G = [](long n) { return dim3((unsigned)((n + TPB - 1) / TPB)); };

  // init: zero output + counters, compute bounds / bin params / cell ids
  k_zero_i32<<<1024, TPB, 0, stream>>>(outI, OS);
  k_zero_i32<<<1, TPB, 0, stream>>>(cnt, 64);
  k_seti<<<1, 1, 0, stream>>>(&cnt[10], N);
  k_init_bounds<<<1, TPB, 0, stream>>>(pi, pf);
  k_bounds<<<G(N), TPB, 0, stream>>>(nodes, N, pi, pf);
  k_params<<<1, 1, 0, stream>>>(pi, pf, N, 64);
  k_assign<<<G(N), TPB, 0, stream>>>(nodes, pf, cell, N, 64);

  const int Mdense[5]   = {262144, 32768, 4096, 512, 64};
  const int rlv[5]      = {64, 32, 16, 8, 4};
  const int capChild[5] = {N, N, 32768, 4096, 512};

  for (int k = 0; k < 5; k++) {
    int M = Mdense[k], rl = rlv[k], cap = capChild[k];
    int nb = (M + TPB - 1) / TPB;
    const int* nChild = (k == 0) ? &cnt[10] : &cnt[k - 1];
    const int* ids    = (k == 0) ? cell : child;
    float* svOut      = (k % 2 == 0) ? svA : svB;
    const float* srcF = (k == 0) ? nodes : ((k % 2 == 0) ? svB : svA);
    int srcStride     = (k == 0) ? 10 : 16;

    // histogram over dense grid
    k_zero_i32<<<G(M), TPB, 0, stream>>>(counts, M);
    k_zero_i32<<<G(M), TPB, 0, stream>>>(cursor, M);
    k_hist<<<G(cap), TPB, 0, stream>>>(ids, nChild, counts);
    // exclusive scan of counts -> starts
    k_scan_block<<<nb, TPB, 0, stream>>>(counts, 0, starts, sums, M);
    k_scan_sums<<<1, 1, 0, stream>>>(sums, nb, (int*)nullptr);
    k_scan_add<<<nb, TPB, 0, stream>>>(starts, sums, M);
    // flag scan -> compact indices + n_level
    k_scan_block<<<nb, TPB, 0, stream>>>(counts, 1, flagsc, sums, M);
    k_scan_sums<<<1, 1, 0, stream>>>(sums, nb, &cnt[k]);
    k_scan_add<<<nb, TPB, 0, stream>>>(flagsc, sums, M);
    k_compact<<<G(M), TPB, 0, stream>>>(counts, flagsc, M, ne, d2c);
    // per-level inter-cell edges
    k_edges<<<G(M), TPB, 0, stream>>>(ne, &cnt[k], counts, d2c, rl, outI,
                                      gEBase, gEEnd, (unsigned int*)&cnt[9]);
    // sort children by segment (counting-sort scatter)
    k_scatter<<<G(cap), TPB, 0, stream>>>(ids, nChild, starts, cursor, d2c, order, segc);
    // assigns output
    k_assigns_out<<<G(cap), TPB, 0, stream>>>(order, segc, nChild, outI, &cnt[6], gABase, gAEnd);
    k_bump<<<1, 1, 0, stream>>>(&cnt[6], nChild, 2);
    // node->cell graph (level 0 only, before buffers are reused)
    if (k == 0)
      k_graph<<<G(N), TPB, 0, stream>>>(cell, counts, starts, order, N, rl,
                                        outI, gGraphEnd, (unsigned int*)&cnt[5]);
    // WMMA segment aggregation
    k_zero_f32<<<G((long)M * 16), TPB, 0, stream>>>(svOut, (long)M * 16);
    k_agg_wmma<<<G(cap), TPB, 0, stream>>>(srcF, srcStride, order, segc, nChild, svOut, &cnt[k]);
    k_finalize_sv<<<G(M), TPB, 0, stream>>>(svOut, &cnt[k]);
    // sv + ids output
    k_sv_out<<<G((long)M * 10), TPB, 0, stream>>>(svOut, &cnt[k], outF, &cnt[7], gSBase, gSEnd);
    k_bump<<<1, 1, 0, stream>>>(&cnt[7], &cnt[k], 10);
    k_ids_out<<<G(M), TPB, 0, stream>>>(ne, &cnt[k], outI, &cnt[8], gIBase, gIEnd);
    k_bump<<<1, 1, 0, stream>>>(&cnt[8], &cnt[k], 1);
    // parent ids for next level
    if (k < 4)
      k_parent<<<G(M), TPB, 0, stream>>>(ne, &cnt[k], rl, child);
  }
}
